// MultiExpertImputer_29764123361485
// MI455X (gfx1250) — matrix operand order repaired and checked by reading, not compiled, not run on hardware
//
#include <hip/hip_runtime.h>
#include <hip/hip_bf16.h>
#include <math.h>

// Problem dims (fixed by the reference)
#define E_ 8
#define B_ 4096
#define D_ 1024
#define H_ 2048
#define EPS_ 1e-5f

typedef __bf16 bf16;
typedef bf16 bvec2  __attribute__((ext_vector_type(2)));
typedef bf16 bvec8  __attribute__((ext_vector_type(8)));
typedef bf16 bvec16 __attribute__((ext_vector_type(16)));
typedef float f32x4 __attribute__((ext_vector_type(4)));
typedef float f32x8 __attribute__((ext_vector_type(8)));

// ---------------------------------------------------------------------------
// Tiled WMMA GEMM:  C[M,N] = act( concatK(A0[:, :K0], A1[:, K0:K]) @ W[K,N] + bias )
// fp32 operands converted to bf16 while staging into double-buffered LDS.
// Macro tile 128x128, 8 waves (2x4), each wave 64x32 = 8 x (16x16x32) WMMA tiles.
// Requires M%128==0, N%128==0, K%32==0, K0%32==0 (true for all uses here).
// ---------------------------------------------------------------------------
template <int RELU>
__global__ __launch_bounds__(256)
void gemm_bf16_wmma(const float* __restrict__ A0, const float* __restrict__ A1,
                    int K0, int K,
                    const float* __restrict__ W, const float* __restrict__ bias,
                    float* __restrict__ C,
                    int lda0, int lda1, int ldw, int ldc)
{
    constexpr int BM = 128, BN = 128, BK = 32, LSTR = BK + 8; // pad vs bank conflicts
    __shared__ bf16 sA[2][BM][LSTR];   // [m][k]
    __shared__ bf16 sB[2][BN][LSTR];   // [n][k]  (W tile stored transposed)

    const int tid  = threadIdx.x;
    const int lane = tid & 31;
    const int wave = tid >> 5;
    const int wm   = wave >> 2;        // 0..1
    const int wn   = wave & 3;         // 0..3
    const int l16  = lane & 15;
    const int hi   = lane >> 4;        // 0 or 1
    const int m_base = blockIdx.x * BM;
    const int n_base = blockIdx.y * BN;

    // Stage A tile (BM x BK): 512 chunks of 8 k-values; 2 f32x4 loads -> 1 b128 store.
    auto stageA = [&](int kt, int buf) {
        #pragma unroll
        for (int it = 0; it < 2; ++it) {
            int c   = tid + it * 256;        // 0..511
            int row = c >> 2;
            int kc  = (c & 3) * 8;
            int gk  = kt + kc;               // gk..gk+7 on one side of the K split
            const float* src = (gk < K0)
                ? (A0 + (size_t)(m_base + row) * lda0 + gk)
                : (A1 + (size_t)(m_base + row) * lda1 + (gk - K0));
            f32x4 v0 = *(const f32x4*)src;
            f32x4 v1 = *(const f32x4*)(src + 4);
            bvec8 o;
            o[0] = (bf16)v0[0]; o[1] = (bf16)v0[1]; o[2] = (bf16)v0[2]; o[3] = (bf16)v0[3];
            o[4] = (bf16)v1[0]; o[5] = (bf16)v1[1]; o[6] = (bf16)v1[2]; o[7] = (bf16)v1[3];
            *(bvec8*)&sA[buf][row][kc] = o;
            if (kt + BK < K) {               // prefetch tile t+2
                int nk = gk + BK;
                const float* nsrc = (nk < K0)
                    ? (A0 + (size_t)(m_base + row) * lda0 + nk)
                    : (A1 + (size_t)(m_base + row) * lda1 + (nk - K0));
                __builtin_prefetch(nsrc, 0, 1);
            }
        }
    };
    // Stage W tile transposed (BK x BN -> [n][k]): per chunk, two coalesced f32x4
    // rows (k, k+1); pack column pairs with v_cvt_pk_bf16_f32 -> 4 b32 stores.
    auto stageB = [&](int kt, int buf) {
        #pragma unroll
        for (int it = 0; it < 2; ++it) {
            int c  = tid + it * 256;         // 0..511
            int kp = c >> 5;                 // 0..15  -> k = 2*kp
            int nc = (c & 31) * 4;           // 0..124
            int k  = kp * 2;
            const float* src = W + (size_t)(kt + k) * ldw + n_base + nc;
            f32x4 r0 = *(const f32x4*)src;
            f32x4 r1 = *(const f32x4*)(src + ldw);
            #pragma unroll
            for (int j = 0; j < 4; ++j) {
                bvec2 p;
                p[0] = (bf16)r0[j];
                p[1] = (bf16)r1[j];
                *(bvec2*)&sB[buf][nc + j][k] = p;
            }
        }
    };

    f32x8 acc[4][2];
    const f32x8 zero = {0.f,0.f,0.f,0.f,0.f,0.f,0.f,0.f};
    #pragma unroll
    for (int i = 0; i < 4; ++i)
        #pragma unroll
        for (int j = 0; j < 2; ++j) acc[i][j] = zero;

    stageA(0, 0);
    stageB(0, 0);
    __syncthreads();

    const int nTiles = K / BK;
    for (int t = 0; t < nTiles; ++t) {
        const int cur = t & 1;
        if (t + 1 < nTiles) {                // fill the other buffer while computing
            stageA((t + 1) * BK, cur ^ 1);
            stageB((t + 1) * BK, cur ^ 1);
        }

        // ---- fragment loads (ISA 16-bit A 16x32 / B 32x16 lane layouts) ----
        bvec16 afrag[4];
        #pragma unroll
        for (int mt = 0; mt < 4; ++mt) {
            int row = wm * 64 + mt * 16 + l16;
            bvec8 lo = *(const bvec8*)&sA[cur][row][hi * 8];        // K {0..7}/{8..15}
            bvec8 hh = *(const bvec8*)&sA[cur][row][hi * 8 + 16];   // K {16..23}/{24..31}
            afrag[mt] = __builtin_shufflevector(lo, hh,
                0,1,2,3,4,5,6,7,8,9,10,11,12,13,14,15);
        }
        bvec16 bfrag[2];
        #pragma unroll
        for (int nt = 0; nt < 2; ++nt) {
            int col = wn * 32 + nt * 16 + l16;
            bvec8 lo = *(const bvec8*)&sB[cur][col][hi * 8];        // K {0..7}/{8..15}
            bvec8 hh = *(const bvec8*)&sB[cur][col][hi * 8 + 16];   // K {16..23}/{24..31}
            bfrag[nt] = __builtin_shufflevector(lo, hh,
                0,1,2,3,4,5,6,7,8,9,10,11,12,13,14,15);
        }
        #pragma unroll
        for (int mt = 0; mt < 4; ++mt)
            #pragma unroll
            for (int nt = 0; nt < 2; ++nt)
                acc[mt][nt] = __builtin_amdgcn_wmma_f32_16x16x32_bf16(
                    false, afrag[mt], false, bfrag[nt],
                    (short)0, acc[mt][nt], false, false);
        __syncthreads();
    }

    // ---- epilogue: bias + activation, fp32 store (C/D lane layout) ----
    #pragma unroll
    for (int mt = 0; mt < 4; ++mt) {
        #pragma unroll
        for (int nt = 0; nt < 2; ++nt) {
            int n = n_base + wn * 32 + nt * 16 + l16;
            float bv = bias ? bias[n] : 0.f;
            #pragma unroll
            for (int r = 0; r < 8; ++r) {
                int m = m_base + wm * 64 + mt * 16 + hi * 8 + r;
                float v = acc[mt][nt][r] + bv;
                if (RELU) v = v > 0.f ? v : 0.f;
                C[(size_t)m * ldc + n] = v;
            }
        }
    }
}

// ---------------------------------------------------------------------------
// Helpers / elementwise kernels
// ---------------------------------------------------------------------------
__device__ inline float wave_reduce_sum(float v) {
    #pragma unroll
    for (int off = 16; off > 0; off >>= 1) v += __shfl_xor(v, off, 32);
    return v;
}

__global__ __launch_bounds__(256)
void zero_kernel(float* __restrict__ p, int n) {
    for (int i = blockIdx.x * 256 + threadIdx.x; i < n; i += gridDim.x * 256)
        p[i] = 0.f;
}

// pos = LN(t + de) in place (t buffer), per-expert gamma/beta. One block per row.
__global__ __launch_bounds__(256)
void add_ln_kernel(float* __restrict__ t, const float* __restrict__ de,
                   const float* __restrict__ g, const float* __restrict__ b)
{
    const int row = blockIdx.x, tid = threadIdx.x;
    __shared__ float red[8];
    float vals[4], s = 0.f;
    #pragma unroll
    for (int i = 0; i < 4; ++i) {
        int d = tid + i * 256;
        vals[i] = t[(size_t)row * D_ + d] + de[(size_t)row * D_ + d];
        s += vals[i];
    }
    s = wave_reduce_sum(s);
    if ((tid & 31) == 0) red[tid >> 5] = s;
    __syncthreads();
    float mu = 0.f;
    #pragma unroll
    for (int w = 0; w < 8; ++w) mu += red[w];
    mu *= (1.f / (float)D_);
    __syncthreads();
    float vs = 0.f;
    #pragma unroll
    for (int i = 0; i < 4; ++i) { float dv = vals[i] - mu; vs += dv * dv; }
    vs = wave_reduce_sum(vs);
    if ((tid & 31) == 0) red[tid >> 5] = vs;
    __syncthreads();
    float var = 0.f;
    #pragma unroll
    for (int w = 0; w < 8; ++w) var += red[w];
    var *= (1.f / (float)D_);
    float rstd = rsqrtf(var + EPS_);
    #pragma unroll
    for (int i = 0; i < 4; ++i) {
        int d = tid + i * 256;
        t[(size_t)row * D_ + d] = g[d] * (vals[i] - mu) * rstd + b[d];
    }
}

// out += sparse[row,e] * LN(gate*cand + (1-gate)*ce). One block per row.
__global__ __launch_bounds__(256)
void combine_ln_kernel(const float* __restrict__ cand, const float* __restrict__ ce,
                       const float* __restrict__ gate,
                       const float* __restrict__ g, const float* __restrict__ b,
                       const float* __restrict__ sparse, int e,
                       float* __restrict__ out)
{
    const int row = blockIdx.x, tid = threadIdx.x;
    __shared__ float red[8];
    const float gt = gate[row];
    const float w  = sparse[(size_t)row * E_ + e];
    float vals[4], s = 0.f;
    #pragma unroll
    for (int i = 0; i < 4; ++i) {
        int d = tid + i * 256;
        vals[i] = gt * cand[(size_t)row * D_ + d] + (1.f - gt) * ce[(size_t)row * D_ + d];
        s += vals[i];
    }
    s = wave_reduce_sum(s);
    if ((tid & 31) == 0) red[tid >> 5] = s;
    __syncthreads();
    float mu = 0.f;
    #pragma unroll
    for (int wv = 0; wv < 8; ++wv) mu += red[wv];
    mu *= (1.f / (float)D_);
    __syncthreads();
    float vs = 0.f;
    #pragma unroll
    for (int i = 0; i < 4; ++i) { float dv = vals[i] - mu; vs += dv * dv; }
    vs = wave_reduce_sum(vs);
    if ((tid & 31) == 0) red[tid >> 5] = vs;
    __syncthreads();
    float var = 0.f;
    #pragma unroll
    for (int wv = 0; wv < 8; ++wv) var += red[wv];
    var *= (1.f / (float)D_);
    float rstd = rsqrtf(var + EPS_);
    #pragma unroll
    for (int i = 0; i < 4; ++i) {
        int d = tid + i * 256;
        float ev = g[d] * (vals[i] - mu) * rstd + b[d];
        out[(size_t)row * D_ + d] += w * ev;
    }
}

// gate[row] = sigmoid(h[row,:] . w2 + b2). One wave per row.
__global__ __launch_bounds__(256)
void gate_dot_kernel(const float* __restrict__ h, const float* __restrict__ w2,
                     const float* __restrict__ b2, float* __restrict__ gate)
{
    const int lane = threadIdx.x & 31;
    const int row  = blockIdx.x * 8 + (threadIdx.x >> 5);
    float s = 0.f;
    for (int k = lane; k < H_; k += 32)
        s += h[(size_t)row * H_ + k] * w2[k];
    s = wave_reduce_sum(s);
    if (lane == 0) gate[row] = 1.f / (1.f + expf(-(s + b2[0])));
}

// scores[row, 0..7] = rh[row,:] @ r_W2 + r_b2. One wave per row, 8 accumulators.
__global__ __launch_bounds__(256)
void router_scores_kernel(const float* __restrict__ rh, const float* __restrict__ W2,
                          const float* __restrict__ b2, float* __restrict__ scores)
{
    const int lane = threadIdx.x & 31;
    const int row  = blockIdx.x * 8 + (threadIdx.x >> 5);
    float acc[E_];
    #pragma unroll
    for (int e = 0; e < E_; ++e) acc[e] = 0.f;
    for (int k = lane; k < H_; k += 32) {
        float rv = rh[(size_t)row * H_ + k];
        const float* wr = W2 + (size_t)k * E_;
        #pragma unroll
        for (int e = 0; e < E_; ++e) acc[e] += rv * wr[e];
    }
    #pragma unroll
    for (int e = 0; e < E_; ++e) {
        float s = wave_reduce_sum(acc[e]);
        if (lane == 0) scores[(size_t)row * E_ + e] = s + b2[e];
    }
}

// Per-row softmax over 8 scores, keep top-2, write dense sparse row.
__global__ __launch_bounds__(256)
void softmax_topk_kernel(const float* __restrict__ scores, float* __restrict__ sparse)
{
    const int row = blockIdx.x * 256 + threadIdx.x;
    if (row >= B_) return;
    float sc[E_], mx = -3.4e38f;
    #pragma unroll
    for (int e = 0; e < E_; ++e) { sc[e] = scores[(size_t)row * E_ + e]; mx = fmaxf(mx, sc[e]); }
    float sum = 0.f;
    #pragma unroll
    for (int e = 0; e < E_; ++e) { sc[e] = expf(sc[e] - mx); sum += sc[e]; }
    float inv = 1.f / sum;
    #pragma unroll
    for (int e = 0; e < E_; ++e) sc[e] *= inv;
    int i1 = 0;
    #pragma unroll
    for (int e = 1; e < E_; ++e) if (sc[e] > sc[i1]) i1 = e;
    int i2 = (i1 == 0) ? 1 : 0;
    #pragma unroll
    for (int e = 0; e < E_; ++e) if (e != i1 && sc[e] > sc[i2]) i2 = e;
    #pragma unroll
    for (int e = 0; e < E_; ++e)
        sparse[(size_t)row * E_ + e] = (e == i1 || e == i2) ? sc[e] : 0.f;
}

// ---------------------------------------------------------------------------
// Host orchestration
// ---------------------------------------------------------------------------
extern "C" void kernel_launch(void* const* d_in, const int* in_sizes, int n_in,
                              void* d_out, int out_size, void* d_ws, size_t ws_size,
                              hipStream_t stream)
{
    const float* de      = (const float*)d_in[0];
    const float* ce      = (const float*)d_in[1];
    const float* dpe_W1  = (const float*)d_in[2];
    const float* dpe_b1  = (const float*)d_in[3];
    const float* dpe_W2  = (const float*)d_in[4];
    const float* dpe_b2  = (const float*)d_in[5];
    const float* dpe_W3  = (const float*)d_in[6];
    const float* dpe_b3  = (const float*)d_in[7];
    const float* dpe_Wo  = (const float*)d_in[8];
    const float* dpe_bo  = (const float*)d_in[9];
    const float* dpe_lng = (const float*)d_in[10];
    const float* dpe_lnb = (const float*)d_in[11];
    const float* fus_W1  = (const float*)d_in[12];
    const float* fus_b1  = (const float*)d_in[13];
    const float* fus_W2  = (const float*)d_in[14];
    const float* fus_b2  = (const float*)d_in[15];
    const float* gate_W1 = (const float*)d_in[16];
    const float* gate_b1 = (const float*)d_in[17];
    const float* gate_W2 = (const float*)d_in[18];
    const float* gate_b2 = (const float*)d_in[19];
    const float* ln2_g   = (const float*)d_in[20];
    const float* ln2_b   = (const float*)d_in[21];
    const float* r_W1    = (const float*)d_in[22];
    const float* r_b1    = (const float*)d_in[23];
    const float* r_W2    = (const float*)d_in[24];
    const float* r_b2    = (const float*)d_in[25];

    float* out    = (float*)d_out;
    float* sparse = out + (size_t)B_ * D_;   // outputs: imputed [B,D] then sparse [B,E]

    float* ws     = (float*)d_ws;
    float* rh     = ws;                                // [B,H]
    float* ha     = rh  + (size_t)B_ * H_;             // [B,H]
    float* hb     = ha  + (size_t)B_ * H_;             // [B,H]
    float* pos    = hb  + (size_t)B_ * H_;             // [B,D]
    float* cand   = pos + (size_t)B_ * D_;             // [B,D]
    float* scores = cand + (size_t)B_ * D_;            // [B,E]
    float* gate   = scores + (size_t)B_ * E_;          // [B]

    auto gemm = [&](const float* A0, const float* A1, int K0, int K,
                    const float* W, const float* bias, float* C,
                    int N, int lda0, int lda1, int ldw, int ldc, bool relu) {
        dim3 grid(B_ / 128, N / 128);
        if (relu)
            gemm_bf16_wmma<1><<<grid, 256, 0, stream>>>(A0, A1, K0, K, W, bias, C,
                                                        lda0, lda1, ldw, ldc);
        else
            gemm_bf16_wmma<0><<<grid, 256, 0, stream>>>(A0, A1, K0, K, W, bias, C,
                                                        lda0, lda1, ldw, ldc);
    };

    // init imputed accumulator
    zero_kernel<<<4096, 256, 0, stream>>>(out, B_ * D_);

    // ---- router (independent of experts) ----
    gemm(de, ce, D_, 2 * D_, r_W1, r_b1, rh, H_, D_, D_, H_, H_, true);
    router_scores_kernel<<<B_ / 8, 256, 0, stream>>>(rh, r_W2, r_b2, scores);
    softmax_topk_kernel<<<(B_ + 255) / 256, 256, 0, stream>>>(scores, sparse);

    // ---- dense expert stack, one expert at a time (ws reuse) ----
    for (int e = 0; e < E_; ++e) {
        const float* W1 = dpe_W1 + (size_t)e * D_ * H_;
        const float* W2 = dpe_W2 + (size_t)e * H_ * H_;
        const float* W3 = dpe_W3 + (size_t)e * H_ * H_;
        const float* Wo = dpe_Wo + (size_t)e * H_ * D_;

        gemm(de, nullptr, D_, D_, W1, dpe_b1 + (size_t)e * H_, ha, H_, D_, 0, H_, H_, true);
        gemm(ha, nullptr, H_, H_, W2, dpe_b2 + (size_t)e * H_, hb, H_, H_, 0, H_, H_, true);
        gemm(hb, nullptr, H_, H_, W3, dpe_b3 + (size_t)e * H_, ha, H_, H_, 0, H_, H_, true);
        gemm(ha, nullptr, H_, H_, Wo, dpe_bo + (size_t)e * D_, pos, D_, H_, 0, D_, D_, false);
        add_ln_kernel<<<B_, 256, 0, stream>>>(pos, de,
                                              dpe_lng + (size_t)e * D_,
                                              dpe_lnb + (size_t)e * D_);
        // fused = concat(ce, pos) handled by split-A GEMM
        gemm(ce, pos, D_, 2 * D_, fus_W1 + (size_t)e * 2 * D_ * H_,
             fus_b1 + (size_t)e * H_, ha, H_, D_, D_, H_, H_, true);
        gemm(ha, nullptr, H_, H_, fus_W2 + (size_t)e * H_ * D_,
             fus_b2 + (size_t)e * D_, cand, D_, H_, 0, D_, D_, false);
        gemm(ce, pos, D_, 2 * D_, gate_W1 + (size_t)e * 2 * D_ * H_,
             gate_b1 + (size_t)e * H_, hb, H_, D_, D_, H_, H_, true);
        gate_dot_kernel<<<B_ / 8, 256, 0, stream>>>(hb, gate_W2 + (size_t)e * H_,
                                                    gate_b2 + e, gate);
        combine_ln_kernel<<<B_, 256, 0, stream>>>(cand, ce, gate,
                                                  ln2_g + (size_t)e * D_,
                                                  ln2_b + (size_t)e * D_,
                                                  sparse, e, out);
    }
}
